// drugcell_graph_23948737642813
// MI455X (gfx1250) — compile-verified, matrix operand order checked by reading, not run on hardware
//
#include <hip/hip_runtime.h>
#include <hip/hip_bf16.h>
#include <math.h>

typedef __attribute__((ext_vector_type(16))) __bf16 v16bf;
typedef __attribute__((ext_vector_type(8)))  float  v8f;

constexpr int kT   = 511;
constexpr int kLeafLo = 255;
constexpr int kNLeaf  = 256;
constexpr int kB   = 512;
constexpr int kNG  = 3008;
constexpr int kG   = 8;
constexpr int kH   = 64;
constexpr int kA   = 64;
constexpr int kHD  = 128;
constexpr int kF   = 64;
constexpr int kIN  = 2 * kH + kG;   // 136
constexpr int kKP  = 160;           // 136 padded to 5 k-frags
constexpr float kEps = 1e-5f;

// d_out layout (floats): aux_out (T,B,1) | term_out (T,B,H) | fout (B,F) | pred (B,1)
constexpr long kToOff   = (long)kT * kB;                   // 261632
constexpr long kFoutOff = kToOff + (long)kT * kB * kH;     // 17006080
constexpr long kPredOff = kFoutOff + (long)kB * kF;        // 17038848

// Fragment-layout helpers (CDNA5 16-bit A/B VGPR layout, 16x32 k-tile):
//   lane = row/col + 16*((kk>>3)&1),  elem = (kk&7) + (kk>=16 ? 8 : 0)
__device__ __forceinline__ int frag_lane16(int rc, int kk) { return rc + 16 * ((kk >> 3) & 1); }
__device__ __forceinline__ int frag_elem(int kk)           { return (kk & 7) + ((kk & 16) ? 8 : 0); }

__device__ __forceinline__ v8f wmma_bf16(v16bf a, v16bf b, v8f c) {
    return __builtin_amdgcn_wmma_f32_16x16x32_bf16(false, a, false, b, (short)0, c, false, false);
}

// ---------------------------------------------------------------- K1: f32 -> bf16
__global__ __launch_bounds__(256) void cvt_bf16_kernel(const float* __restrict__ src,
                                                       __bf16* __restrict__ dst, long n) {
    long i = (long)blockIdx.x * 256 + threadIdx.x;
    if (i < n) dst[i] = (__bf16)src[i];
}

// ---------------------------------------------------------------- K2: grouped WMMA GEMM
// term_gene[t, b, g] = gene[b, :] @ W_gene[t, :, g] + b_gene[t, g]
__global__ __launch_bounds__(256) void termgene_wmma_kernel(
        const __bf16* __restrict__ gbf, const float* __restrict__ Wg,
        const float* __restrict__ bg, float* __restrict__ tg,
        __bf16* __restrict__ tgbf) {
    __shared__ __align__(32) __bf16       sA[2][32][16];     // 2 k-frags, frag layout (2 KB)
    __shared__ __align__(32) unsigned int sB[8][2][32][8];   // per-term, packed bf16 pairs (16 KB)

    const int tid  = threadIdx.x;
    const int lane = tid & 31;
    const int w    = tid >> 5;                 // wave id = term slot
    const int m0   = blockIdx.x * 16;
    const int grp  = blockIdx.y;

    int t = grp * 8 + w;
    const bool valid = (t < kT);
    if (!valid) t = kT - 1;                    // clamp; wave still runs (barriers, EXEC all-1s)
    const long tbase = (long)t * kNG * kG;

    // one-time zero fill of B-fragment lanes n>=8 (padding columns); never rewritten
    {
        unsigned int* sbf = &sB[0][0][0][0];
        for (int i = tid; i < 8 * 2 * 32 * 8; i += 256) {
            int ln = (i >> 3) & 31;
            if ((ln & 15) >= kG) sbf[i] = 0u;
        }
    }

    v8f acc = {0.f, 0.f, 0.f, 0.f, 0.f, 0.f, 0.f, 0.f};

    for (int k0 = 0; k0 < kNG; k0 += 64) {
        __syncthreads();
        // stage A: 16 rows x 64 k -> fragment layout; 4 consecutive bf16 per thread
        {
            int g4 = tid * 4;                        // 1024 elems, exactly 1 group/thread
            int r = g4 >> 6, c = g4 & 63;
            unsigned long long v =
                *(const unsigned long long*)(gbf + (long)(m0 + r) * kNG + k0 + c);
            int frag = c >> 5, kk = c & 31;
            *(unsigned long long*)&sA[frag][frag_lane16(r, kk)][frag_elem(kk)] = v;
        }
        // stage B: each term 64 k-rows x 8 g; thread (term w, lane l) handles rows 2l, 2l+1
        {
            int tw = tid >> 5, l = tid & 31;
            int tt = grp * 8 + tw; if (tt >= kT) tt = kT - 1;
            const float4* src =
                (const float4*)(Wg + (long)tt * kNG * kG + (long)(k0 + 2 * l) * kG);
            float4 r0a = src[0], r0b = src[1], r1a = src[2], r1b = src[3];
            float row0[8] = {r0a.x, r0a.y, r0a.z, r0a.w, r0b.x, r0b.y, r0b.z, r0b.w};
            float row1[8] = {r1a.x, r1a.y, r1a.z, r1a.w, r1b.x, r1b.y, r1b.z, r1b.w};
            int kl = 2 * l;                          // even local k
            int frag = kl >> 5, kk = kl & 31;
            int ln16 = 16 * ((kk >> 3) & 1);
            int p = frag_elem(kk) >> 1;              // pair index
            #pragma unroll
            for (int g = 0; g < kG; ++g) {
                union { __bf16 h[2]; unsigned int u; } pk;
                pk.h[0] = (__bf16)row0[g];
                pk.h[1] = (__bf16)row1[g];
                sB[tw][frag][g + ln16][p] = pk.u;
            }
        }
        if (k0 + 64 < kNG)
            __builtin_prefetch(Wg + tbase + (long)(k0 + 64) * kG, 0, 3);
        __syncthreads();

        v16bf a0 = *(const v16bf*)&sA[0][lane][0];
        v16bf b0 = *(const v16bf*)&sB[w][0][lane][0];
        v16bf a1 = *(const v16bf*)&sA[1][lane][0];
        v16bf b1 = *(const v16bf*)&sB[w][1][lane][0];
        acc = wmma_bf16(a0, b0, acc);
        acc = wmma_bf16(a1, b1, acc);
    }

    const int n = lane & 15;
    if (valid && n < kG) {
        float bias = bg[t * kG + n];
        #pragma unroll
        for (int r = 0; r < 8; ++r) {
            int mg = m0 + r + ((lane >= 16) ? 8 : 0);   // C-layout: VGPR r -> M=r / r+8
            float v = acc[r] + bias;
            tg[(long)t * kB * kG + (long)mg * kG + n] = v;
            tgbf[(long)t * kB * kG + (long)mg * kG + n] = (__bf16)v;
        }
    }
}

// ---------------------------------------------------------------- K3: leaf terms + batch-BN
__global__ __launch_bounds__(256) void leaf_kernel(
        const float* __restrict__ tg, const float* __restrict__ Wl,
        const float* __restrict__ bl, const float* __restrict__ gamma,
        const float* __restrict__ beta, float* out, __bf16* __restrict__ tobf) {
    __shared__ float sW[kG][kH];
    __shared__ float sSum[4][kH], sSq[4][kH], sScale[kH], sShift[kH];
    const int l = blockIdx.x;
    const int t = kLeafLo + l;
    const int tid = threadIdx.x;
    for (int i = tid; i < kG * kH; i += 256) sW[i >> 6][i & 63] = Wl[(long)l * kG * kH + i];
    __syncthreads();

    const int j = tid & 63, sg = tid >> 6;
    const float bias = bl[l * kH + j];
    float* to = out + kToOff + (long)t * kB * kH;
    __bf16* tob = tobf + (long)t * kB * kH;
    const float* tgr = tg + (long)t * kB * kG;
    float s = 0.f, ss = 0.f;
    for (int b = sg * 128; b < sg * 128 + 128; ++b) {
        const float* x = tgr + (long)b * kG;
        float acc = bias;
        #pragma unroll
        for (int g = 0; g < kG; ++g) acc += x[g] * sW[g][j];
        float h = tanhf(acc);
        to[(long)b * kH + j] = h;
        s += h; ss += h * h;
    }
    sSum[sg][j] = s; sSq[sg][j] = ss;
    __syncthreads();
    if (tid < kH) {
        float st = 0.f, sq = 0.f;
        #pragma unroll
        for (int q = 0; q < 4; ++q) { st += sSum[q][tid]; sq += sSq[q][tid]; }
        float mean = st * (1.0f / kB);
        float var  = sq * (1.0f / kB) - mean * mean;
        float sc   = gamma[t * kH + tid] * rsqrtf(var + kEps);
        sScale[tid] = sc;
        sShift[tid] = beta[t * kH + tid] - mean * sc;
    }
    __syncthreads();
    for (int b = sg * 128; b < sg * 128 + 128; ++b) {
        float h = to[(long)b * kH + j] * sScale[j] + sShift[j];
        to[(long)b * kH + j] = h;
        tob[(long)b * kH + j] = (__bf16)h;
    }
}

// ---------------------------------------------------------------- K4: internal level, WMMA
// h[t] (512 x 64) = tanh([child0|child1|genes] (512 x 136) @ W_int[t] (136 x 64) + b), then batch-BN.
// One block per term, 8 waves; K padded 136->160 (5 frags); batch in 8 chunks of 64 rows.
__global__ __launch_bounds__(256) void internal_wmma_kernel(
        const __bf16* __restrict__ tobf, const __bf16* __restrict__ tgbf,
        const __bf16* __restrict__ wibf, const float* __restrict__ bi,
        const float* __restrict__ gamma, const float* __restrict__ beta,
        float* out, __bf16* tobf_out, int lo) {
    constexpr int NF = kKP / 32;  // 5
    __shared__ __align__(32) __bf16 sB[4][NF][32][16];   // 4 N-tiles (20 KB)
    __shared__ __align__(32) __bf16 sA[4][NF][32][16];   // 4 M-tiles per chunk (20 KB)
    __shared__ float sSum[kH], sSq[kH], sScale[kH], sShift[kH];

    const int t = lo + blockIdx.x;
    const int c0 = 2 * t + 1, c1 = 2 * t + 2;
    const int tid = threadIdx.x, lane = tid & 31, w = tid >> 5;
    const __bf16* x0 = tobf + (long)c0 * kB * kH;   // child0: 64 bf16 per row
    const __bf16* x1 = tobf + (long)c1 * kB * kH;   // child1
    const __bf16* xg = tgbf + (long)t * kB * kG;    // genes: 8 bf16 per row

    for (int i = tid; i < kH; i += 256) { sSum[i] = 0.f; sSq[i] = 0.f; }

    // stage B: W_int[t] (136 x 64) zero-padded to 160 rows, fragment layout
    for (int g4 = tid * 4; g4 < kKP * kH; g4 += 256 * 4) {
        int k = g4 >> 6, n4 = g4 & 63;
        union { unsigned long long u; __bf16 h[4]; } ld;
        ld.u = (k < kIN) ? *(const unsigned long long*)(wibf + ((long)t * kIN + k) * kH + n4)
                         : 0ULL;
        int frag = k >> 5, kk = k & 31;
        int el = frag_elem(kk), ln16 = 16 * ((kk >> 3) & 1);
        #pragma unroll
        for (int q = 0; q < 4; ++q) {
            int n = n4 + q;
            sB[n >> 4][frag][(n & 15) + ln16][el] = ld.h[q];
        }
    }

    const int myMt = w & 3;          // wave's M-tile within chunk
    const int nt0  = (w >> 2) * 2;   // wave's first N-tile
    float* to = out + kToOff + (long)t * kB * kH;

    for (int chunk = 0; chunk < 8; ++chunk) {
        __syncthreads();
        // stage A: 64 batch rows x 160 k; a group of 4 k never straddles a source boundary
        for (int g4 = tid * 4; g4 < 64 * kKP; g4 += 256 * 4) {
            int r = g4 / kKP, c = g4 - r * kKP;
            int b = chunk * 64 + r;
            unsigned long long v;
            if (c < 64)        v = *(const unsigned long long*)(x0 + (long)b * kH + c);
            else if (c < 128)  v = *(const unsigned long long*)(x1 + (long)b * kH + (c - 64));
            else if (c < kIN)  v = *(const unsigned long long*)(xg + (long)b * kG + (c - 128));
            else               v = 0ULL;
            int mt = r >> 4, rr = r & 15;
            int frag = c >> 5, kk = c & 31;
            *(unsigned long long*)&sA[mt][frag][frag_lane16(rr, kk)][frag_elem(kk)] = v;
        }
        __syncthreads();
        #pragma unroll
        for (int u = 0; u < 2; ++u) {
            int nt = nt0 + u;
            v8f acc = {0.f, 0.f, 0.f, 0.f, 0.f, 0.f, 0.f, 0.f};
            #pragma unroll
            for (int kf = 0; kf < NF; ++kf) {
                v16bf av = *(const v16bf*)&sA[myMt][kf][lane][0];
                v16bf bv = *(const v16bf*)&sB[nt][kf][lane][0];
                acc = wmma_bf16(av, bv, acc);
            }
            int j = nt * 16 + (lane & 15);
            float bias = bi[t * kH + j];
            float ps = 0.f, pss = 0.f;
            #pragma unroll
            for (int r = 0; r < 8; ++r) {
                int b = chunk * 64 + myMt * 16 + r + ((lane >= 16) ? 8 : 0);
                float h = tanhf(acc[r] + bias);
                to[(long)b * kH + j] = h;
                ps += h; pss += h * h;
            }
            atomicAdd(&sSum[j], ps);   // ds_add_f32
            atomicAdd(&sSq[j], pss);
        }
    }
    __syncthreads();
    if (tid < kH) {
        float mean = sSum[tid] * (1.0f / kB);
        float var  = sSq[tid] * (1.0f / kB) - mean * mean;
        float sc   = gamma[t * kH + tid] * rsqrtf(var + kEps);
        sScale[tid] = sc;
        sShift[tid] = beta[t * kH + tid] - mean * sc;
    }
    __threadfence();
    __syncthreads();
    __bf16* tob = tobf_out + (long)t * kB * kH;
    for (long i = tid; i < (long)kB * kH; i += 256) {
        int j = (int)(i & 63);
        float h = to[i] * sScale[j] + sShift[j];
        to[i] = h;
        tob[i] = (__bf16)h;
    }
}

// ---------------------------------------------------------------- K5: aux heads
__global__ __launch_bounds__(256) void aux_kernel(
        const float* __restrict__ w1, const float* __restrict__ b1,
        const float* __restrict__ w2, const float* __restrict__ b2, float* out) {
    long idx = (long)blockIdx.x * 256 + threadIdx.x;
    if (idx >= (long)kT * kB) return;
    int t = (int)(idx / kB);
    const float* x = out + kToOff + idx * kH;
    const float* wv = w1 + (long)t * kH;
    float acc = 0.f;
    #pragma unroll 8
    for (int h = 0; h < kH; ++h) acc += x[h] * wv[h];
    float a1 = tanhf(acc + b1[t]);
    out[idx] = a1 * w2[t] + b2[t];
}

// ---------------------------------------------------------------- K6: GCN batched WMMA GEMM
template <int KDIV32, bool RELU>
__global__ __launch_bounds__(256) void gcn_wmma_kernel(
        const __bf16* __restrict__ Abase, long aStride,
        const __bf16* __restrict__ Bbase, long bStride,
        __bf16* __restrict__ Out) {
    constexpr int K = KDIV32 * 32;
    __shared__ __align__(32) __bf16 sA[KDIV32][32][16];
    __shared__ __align__(32) __bf16 sB[8][KDIV32][32][16];

    const int tid = threadIdx.x, lane = tid & 31, w = tid >> 5;
    const int m0 = blockIdx.x * 16;
    const int b  = blockIdx.y;
    const __bf16* Ab = Abase + (long)b * aStride;
    const __bf16* Bb = Bbase + (long)b * bStride;

    for (int g4 = tid * 4; g4 < 16 * K; g4 += 256 * 4) {
        int r = g4 / K, c = g4 % K;
        unsigned long long v = *(const unsigned long long*)(Ab + (long)(m0 + r) * K + c);
        int frag = c >> 5, kk = c & 31;
        *(unsigned long long*)&sA[frag][frag_lane16(r, kk)][frag_elem(kk)] = v;
    }
    for (int g4 = tid * 4; g4 < K * 128; g4 += 256 * 4) {
        int k = g4 >> 7, n4 = g4 & 127;
        union { unsigned long long u; __bf16 h[4]; } ld;
        ld.u = *(const unsigned long long*)(Bb + (long)k * 128 + n4);
        int frag = k >> 5, kk = k & 31;
        int el = frag_elem(kk);
        int ln16 = 16 * ((kk >> 3) & 1);
        #pragma unroll
        for (int q = 0; q < 4; ++q) {
            int n = n4 + q;
            sB[n >> 4][frag][(n & 15) + ln16][el] = ld.h[q];
        }
    }
    __syncthreads();

    v8f acc = {0.f, 0.f, 0.f, 0.f, 0.f, 0.f, 0.f, 0.f};
    #pragma unroll
    for (int kf = 0; kf < KDIV32; ++kf) {
        v16bf av = *(const v16bf*)&sA[kf][lane][0];
        v16bf bv = *(const v16bf*)&sB[w][kf][lane][0];
        acc = wmma_bf16(av, bv, acc);
    }

    const int col = w * 16 + (lane & 15);
    #pragma unroll
    for (int r = 0; r < 8; ++r) {
        int mg = m0 + r + ((lane >= 16) ? 8 : 0);
        float v = acc[r];
        if (RELU) v = fmaxf(v, 0.f);
        Out[((long)b * kA + mg) * kHD + col] = (__bf16)v;
    }
}

// ---------------------------------------------------------------- K7: max-pool over atoms (bf16 -> f32)
__global__ __launch_bounds__(256) void gcn_pool_kernel(
        const __bf16* __restrict__ y, float* __restrict__ dout) {
    long idx = (long)blockIdx.x * 256 + threadIdx.x;   // b*HD
    int o = (int)(idx & (kHD - 1));
    int b = (int)(idx >> 7);
    const __bf16* yb = y + (long)b * kA * kHD + o;
    float m = (float)yb[0];
    #pragma unroll 8
    for (int a = 1; a < kA; ++a) m = fmaxf(m, (float)yb[(long)a * kHD]);
    dout[idx] = m;
}

// ---------------------------------------------------------------- K8: final head (single workgroup)
__global__ __launch_bounds__(256) void final_kernel(
        const float* __restrict__ drug, const float* __restrict__ Wf,
        const float* __restrict__ bfv, const float* __restrict__ gamma,
        const float* __restrict__ beta, const float* __restrict__ afw,
        const float* __restrict__ afb, const float* __restrict__ ow,
        const float* __restrict__ ob, float* out) {
    constexpr int IN = kH + kHD;   // 192
    __shared__ float sW[IN * kF];
    __shared__ float sSum[4][kF], sSq[4][kF], sScale[kF], sShift[kF];
    const int tid = threadIdx.x;
    for (int i = tid; i < IN * kF; i += 256) sW[i] = Wf[i];
    __syncthreads();

    const int j = tid & 63, sg = tid >> 6;
    const float* root = out + kToOff;           // term 0 output (B,H)
    float* fo = out + kFoutOff;
    const float bias = bfv[j];
    float s = 0.f, ss = 0.f;
    for (int b = sg; b < kB; b += 4) {
        float acc = bias;
        const float* xr = root + (long)b * kH;
        for (int i = 0; i < kH; ++i)  acc += xr[i] * sW[i * kF + j];
        const float* dr = drug + (long)b * kHD;
        for (int i = 0; i < kHD; ++i) acc += dr[i] * sW[(kH + i) * kF + j];
        float h = tanhf(acc);
        fo[(long)b * kF + j] = h;
        s += h; ss += h * h;
    }
    sSum[sg][j] = s; sSq[sg][j] = ss;
    __syncthreads();
    if (tid < kF) {
        float st = 0.f, sq = 0.f;
        #pragma unroll
        for (int q = 0; q < 4; ++q) { st += sSum[q][tid]; sq += sSq[q][tid]; }
        float mean = st * (1.0f / kB);
        float var  = sq * (1.0f / kB) - mean * mean;
        float sc   = gamma[tid] * rsqrtf(var + kEps);
        sScale[tid] = sc;
        sShift[tid] = beta[tid] - mean * sc;
    }
    __syncthreads();
    for (int b = sg; b < kB; b += 4) {
        float h = fo[(long)b * kF + j];
        fo[(long)b * kF + j] = h * sScale[j] + sShift[j];
    }
    __threadfence();
    __syncthreads();
    const float w2v = ow[0], b2v = ob[0], ab = afb[0];
    for (int b = tid; b < kB; b += 256) {
        float acc = 0.f;
        for (int i = 0; i < kF; ++i) acc += fo[(long)b * kF + i] * afw[i];
        out[kPredOff + b] = tanhf(acc + ab) * w2v + b2v;
    }
}

// ---------------------------------------------------------------- host launcher
extern "C" void kernel_launch(void* const* d_in, const int* in_sizes, int n_in,
                              void* d_out, int out_size, void* d_ws, size_t ws_size,
                              hipStream_t stream) {
    (void)in_sizes; (void)n_in; (void)out_size; (void)ws_size;
    const float* gene  = (const float*)d_in[0];
    const float* drugi = (const float*)d_in[1];
    const float* graph = (const float*)d_in[2];
    const float* Wg    = (const float*)d_in[3];
    const float* bg    = (const float*)d_in[4];
    const float* Wl    = (const float*)d_in[5];
    const float* bl    = (const float*)d_in[6];
    const float* Wi    = (const float*)d_in[7];
    const float* bi    = (const float*)d_in[8];
    const float* gamma = (const float*)d_in[9];
    const float* beta  = (const float*)d_in[10];
    const float* aw1   = (const float*)d_in[11];
    const float* ab1   = (const float*)d_in[12];
    const float* aw2   = (const float*)d_in[13];
    const float* ab2   = (const float*)d_in[14];
    const float* Wg1   = (const float*)d_in[15];
    const float* Wg2   = (const float*)d_in[16];
    const float* Wg3   = (const float*)d_in[17];
    const float* Wf    = (const float*)d_in[18];
    const float* bfin  = (const float*)d_in[19];
    const float* fgam  = (const float*)d_in[20];
    const float* fbet  = (const float*)d_in[21];
    const float* afw   = (const float*)d_in[22];
    const float* afb   = (const float*)d_in[23];
    const float* ow    = (const float*)d_in[24];
    const float* ob    = (const float*)d_in[25];
    float* out = (float*)d_out;

    // workspace layout (all sizes multiples of 8 bytes)
    char* ws = (char*)d_ws;
    constexpr long kGbfBytes = (long)kB * kNG * 2;             // 3.08 MB
    constexpr long kTgElems  = (long)kT * kB * kG;             // 2.09 M
    constexpr long kToElems  = (long)kT * kB * kH;             // 16.7 M
    constexpr long kWiElems  = (long)(kT - kNLeaf) * kIN * kH; // 255*136*64
    constexpr long kXElems   = (long)kB * kA * kHD;            // 4 M
    constexpr long kGrElems  = (long)kB * kA * kA;             // 2 M
    constexpr long kWElems   = (long)kHD * kHD;                // 16 K
    __bf16* gbf     = (__bf16*)ws;   ws += kGbfBytes;
    float*  tg      = (float*)ws;    ws += kTgElems * 4;
    __bf16* tgbf    = (__bf16*)ws;   ws += kTgElems * 2;
    __bf16* tobf    = (__bf16*)ws;   ws += kToElems * 2;
    __bf16* wibf    = (__bf16*)ws;   ws += kWiElems * 2;
    __bf16* dibf    = (__bf16*)ws;   ws += kXElems * 2;
    __bf16* grbf    = (__bf16*)ws;   ws += kGrElems * 2;
    __bf16* w1bf    = (__bf16*)ws;   ws += kWElems * 2;
    __bf16* w2bf    = (__bf16*)ws;   ws += kWElems * 2;
    __bf16* w3bf    = (__bf16*)ws;   ws += kWElems * 2;
    __bf16* tmpbf   = (__bf16*)ws;   ws += kXElems * 2;
    __bf16* layerbf = (__bf16*)ws;   ws += kXElems * 2;
    float*  ddrug   = (float*)ws;

    // 1) bf16 conversions
    long ngene = (long)kB * kNG;
    cvt_bf16_kernel<<<dim3((unsigned)((ngene + 255) / 256)), 256, 0, stream>>>(gene, gbf, ngene);
    cvt_bf16_kernel<<<dim3((unsigned)((kWiElems + 255) / 256)), 256, 0, stream>>>(Wi, wibf, kWiElems);
    cvt_bf16_kernel<<<dim3((unsigned)((kXElems + 255) / 256)), 256, 0, stream>>>(drugi, dibf, kXElems);
    cvt_bf16_kernel<<<dim3((unsigned)((kGrElems + 255) / 256)), 256, 0, stream>>>(graph, grbf, kGrElems);
    cvt_bf16_kernel<<<dim3((unsigned)((kWElems + 255) / 256)), 256, 0, stream>>>(Wg1, w1bf, kWElems);
    cvt_bf16_kernel<<<dim3((unsigned)((kWElems + 255) / 256)), 256, 0, stream>>>(Wg2, w2bf, kWElems);
    cvt_bf16_kernel<<<dim3((unsigned)((kWElems + 255) / 256)), 256, 0, stream>>>(Wg3, w3bf, kWElems);

    // 2) grouped WMMA GEMM: term_gene (f32 + bf16 mirror)
    termgene_wmma_kernel<<<dim3(kB / 16, 64), 256, 0, stream>>>(gbf, Wg, bg, tg, tgbf);

    // 3) leaves + batch BN (f32 out + bf16 mirror)
    leaf_kernel<<<dim3(kNLeaf), 256, 0, stream>>>(tg, Wl, bl, gamma, beta, out, tobf);

    // 4) internal levels bottom-up, WMMA GEMM per term
    for (int d = 7; d >= 0; --d) {
        int lo = (1 << d) - 1, n = 1 << d;
        internal_wmma_kernel<<<dim3(n), 256, 0, stream>>>(tobf, tgbf, wibf, bi, gamma, beta,
                                                          out, tobf, lo);
    }

    // 5) aux heads for every term
    aux_kernel<<<dim3((unsigned)(((long)kT * kB + 255) / 256)), 256, 0, stream>>>(aw1, ab1, aw2, ab2, out);

    // 6) GCN drug branch: 3 x (XW wmma, adj wmma + ReLU), all bf16 activations
    dim3 ggrid(kA / 16, kB);
    gcn_wmma_kernel<4, false><<<ggrid, 256, 0, stream>>>(dibf,    (long)kA * kHD, w1bf,  0L,             tmpbf);
    gcn_wmma_kernel<2, true ><<<ggrid, 256, 0, stream>>>(grbf,    (long)kA * kA,  tmpbf, (long)kA * kHD, layerbf);
    gcn_wmma_kernel<4, false><<<ggrid, 256, 0, stream>>>(layerbf, (long)kA * kHD, w2bf,  0L,             tmpbf);
    gcn_wmma_kernel<2, true ><<<ggrid, 256, 0, stream>>>(grbf,    (long)kA * kA,  tmpbf, (long)kA * kHD, layerbf);
    gcn_wmma_kernel<4, false><<<ggrid, 256, 0, stream>>>(layerbf, (long)kA * kHD, w3bf,  0L,             tmpbf);
    gcn_wmma_kernel<2, true ><<<ggrid, 256, 0, stream>>>(grbf,    (long)kA * kA,  tmpbf, (long)kA * kHD, layerbf);

    // 7) max-pool over atoms
    gcn_pool_kernel<<<dim3((unsigned)((long)kB * kHD / 256)), 256, 0, stream>>>(layerbf, ddrug);

    // 8) final head: BN over batch + aux + pred (single workgroup)
    final_kernel<<<dim3(1), 256, 0, stream>>>(ddrug, Wf, bfin, fgam, fbet, afw, afb, ow, ob, out);
}
